// NMT_64046552318108
// MI455X (gfx1250) — matrix operand order repaired and checked by reading, not compiled, not run on hardware
//
#include <hip/hip_runtime.h>
#include <hip/hip_bf16.h>
#include <math.h>

// ---------------------------------------------------------------------------
// Problem constants (from reference)
// ---------------------------------------------------------------------------
#define B_  32
#define S_  64
#define T_  64
#define TM1 63      // T-1 decode steps
#define E_  512
#define H_  1024
#define V_  32000
#define KCAT (E_ + H_)        // 1536
#define G4H  (4 * H_)         // 4096
#define H2   (2 * H_)         // 2048

typedef __bf16 bf16_t;
typedef __bf16 v16bf __attribute__((ext_vector_type(16)));
typedef __bf16 v8bf  __attribute__((ext_vector_type(8)));
typedef float  v8f   __attribute__((ext_vector_type(8)));

#define CAT16(lo, hi) __builtin_shufflevector(lo, hi, 0,1,2,3,4,5,6,7,8,9,10,11,12,13,14,15)

// ---------------------------------------------------------------------------
// Elementwise helpers
// ---------------------------------------------------------------------------
__global__ void k_f2b(const float* __restrict__ src, bf16_t* __restrict__ dst, int n) {
    int i = blockIdx.x * blockDim.x + threadIdx.x;
    if (i < n) dst[i] = (bf16_t)src[i];
}

// Tiled f32 -> bf16 transpose: src (R x C) f32, dst[(c)*dstStride + dstOff + r] = src[r][c]
// R, C multiples of 32. blockDim (32, 8), grid (C/32, R/32).
__global__ void k_f2bT(const float* __restrict__ src, bf16_t* __restrict__ dst,
                       int R, int C, int dstStride, int dstOff) {
    __shared__ float tile[32][33];
    int c0 = blockIdx.x << 5, r0 = blockIdx.y << 5;
    int tx = threadIdx.x, ty = threadIdx.y;
#pragma unroll
    for (int i = ty; i < 32; i += 8)
        tile[i][tx] = src[(size_t)(r0 + i) * C + c0 + tx];
    __syncthreads();
#pragma unroll
    for (int i = ty; i < 32; i += 8)
        dst[(size_t)(c0 + i) * dstStride + dstOff + r0 + tx] = (bf16_t)tile[tx][i];
}

// xs[b][t][e] = bf16(emb[target[b][t]][e]),  t in [0, T-1)
__global__ void k_gather_emb(const int* __restrict__ target,
                             const float* __restrict__ emb,
                             bf16_t* __restrict__ xs, int total) {
    int i = blockIdx.x * blockDim.x + threadIdx.x;
    if (i >= total) return;
    int e  = i % E_;
    int bt = i / E_;
    int t  = bt % TM1;
    int b  = bt / TM1;
    int tok = target[b * T_ + t];
    xs[i] = (bf16_t)emb[(size_t)tok * E_ + e];
}

// A_cat (32 x 1536) = [ x_t (32x512) | bf16(h) (32x1024) ]
__global__ void k_assemble(const bf16_t* __restrict__ xs, int t,
                           const float* __restrict__ h,
                           bf16_t* __restrict__ A_cat) {
    int i = blockIdx.x * blockDim.x + threadIdx.x;   // 32*1536
    if (i >= B_ * KCAT) return;
    int col = i % KCAT;
    int b   = i / KCAT;
    if (col < E_)
        A_cat[i] = xs[((size_t)b * TM1 + t) * E_ + col];
    else
        A_cat[i] = (bf16_t)h[b * H_ + (col - E_)];
}

__device__ __forceinline__ float sigf(float x) { return 1.0f / (1.0f + __expf(-x)); }

// LSTM cell elementwise: gates (32x4096) -> c,h (32x1024); also hc[:,0:1024]=bf16(h)
__global__ void k_lstm(const float* __restrict__ gates,
                       float* __restrict__ c, float* __restrict__ h,
                       bf16_t* __restrict__ hc) {
    int i = blockIdx.x * blockDim.x + threadIdx.x;   // 32*1024
    if (i >= B_ * H_) return;
    int j = i % H_;
    int b = i / H_;
    const float* g = gates + (size_t)b * G4H;
    float ig = sigf(g[j]);
    float fg = sigf(g[H_ + j]);
    float gg = tanhf(g[2 * H_ + j]);
    float og = sigf(g[3 * H_ + j]);
    float cn = fg * c[i] + ig * gg;
    c[i] = cn;
    float hn = og * tanhf(cn);
    h[i] = hn;
    hc[(size_t)b * H2 + j] = (bf16_t)hn;
}

// Attention for one batch b per block (blockDim = 256)
__global__ void k_attn(const float* __restrict__ enc_proj,
                       const float* __restrict__ h,
                       bf16_t* __restrict__ hc) {
    int b = blockIdx.x;
    const float* ep = enc_proj + (size_t)b * S_ * H_;
    const float* hb = h + (size_t)b * H_;
    __shared__ float sc[S_];
    __shared__ float at[S_];
    int tid = threadIdx.x;
    if (tid < S_) {
        const float* row = ep + (size_t)tid * H_;
        float acc = 0.f;
        for (int j = 0; j < H_; ++j) acc += row[j] * hb[j];
        sc[tid] = acc;
    }
    __syncthreads();
    float mx = -1e30f;
    for (int s = 0; s < S_; ++s) mx = fmaxf(mx, sc[s]);
    if (tid < S_) at[tid] = __expf(sc[tid] - mx);
    __syncthreads();
    float denom = 0.f;
    for (int s = 0; s < S_; ++s) denom += at[s];
    float inv = 1.0f / denom;
    for (int j = tid; j < H_; j += blockDim.x) {
        float acc = 0.f;
        for (int s = 0; s < S_; ++s) acc += at[s] * ep[(size_t)s * H_ + j];
        hc[(size_t)b * H2 + H_ + j] = (bf16_t)(acc * inv);
    }
}

// ---------------------------------------------------------------------------
// bf16 WMMA GEMM:  C(MxN) = A(MxK) @ B(KxN) + bias, optional tanh.
//   A   : row-major (M x K) bf16
//   BT  : B transposed, row-major (N x K) bf16  -> B fragment = 2 contiguous
//         b128 loads per lane, mirroring the A fragment.
// Block = 128 threads = 4 waves. Wave tile = 32 x 64 (2 M-tiles x 4 N-tiles,
// B fragment shared across the 2 M-tiles). Block tile = 32 x 256.
// Grids are exact: requires M % 32 == 0, N % 256 == 0, K % 32 == 0.
// Fragment layouts per CDNA5 ISA 7.12.2 (no guards -> EXEC all-ones).
// ---------------------------------------------------------------------------
__global__ void k_gemm_wmma(const bf16_t* __restrict__ A, int lda,
                            const bf16_t* __restrict__ BT,
                            const float* __restrict__ bias,
                            float* __restrict__ Cf, int ldcf,
                            bf16_t* __restrict__ Cb, int ldcb,
                            int K, int act) {
    const int lane = threadIdx.x & 31;
    const int wave = threadIdx.x >> 5;
    const int half = lane >> 4;       // 0: lanes 0-15, 1: lanes 16-31
    const int l16  = lane & 15;
    const int m0   = blockIdx.y << 5;                 // 32 rows per block
    const int nw   = (blockIdx.x * 4 + wave) << 6;    // 64 cols per wave

    v8f acc[2][4] = {};
    // A fragment rows: lanes 0-15 and 16-31 both hold row m0(+16)+l16;
    // per-lane K slices: {k0+8*half .. +7} and {k0+16+8*half .. +7}.
    const bf16_t* a0p = A + (size_t)(m0 + l16) * lda + 8 * half;
    const bf16_t* a1p = a0p + (size_t)16 * lda;

    for (int k0 = 0; k0 < K; k0 += 32) {
        v8bf alo0 = *(const v8bf*)(a0p + k0);
        v8bf ahi0 = *(const v8bf*)(a0p + k0 + 16);
        v8bf alo1 = *(const v8bf*)(a1p + k0);
        v8bf ahi1 = *(const v8bf*)(a1p + k0 + 16);
        v16bf a0 = CAT16(alo0, ahi0);
        v16bf a1 = CAT16(alo1, ahi1);
#pragma unroll
        for (int s = 0; s < 4; ++s) {
            // B fragment: lane holds BT[n0+l16][k0+8*half+j] and [k0+16+8*half+j], j=0..7 (mirrors A)
            const bf16_t* bp = BT + (size_t)(nw + s * 16 + l16) * K + k0 + 8 * half;
            v8bf blo = *(const v8bf*)bp;
            v8bf bhi = *(const v8bf*)(bp + 16);
            v16bf bfr = CAT16(blo, bhi);
            __builtin_prefetch(bp + 32, 0, 0);   // speculative: next K chunk of BT stream
            acc[0][s] = __builtin_amdgcn_wmma_f32_16x16x32_bf16(
                false, a0, false, bfr, (short)0, acc[0][s], false, false);
            acc[1][s] = __builtin_amdgcn_wmma_f32_16x16x32_bf16(
                false, a1, false, bfr, (short)0, acc[1][s], false, false);
        }
    }

    // Epilogue: acc[mt][s][r] -> row m0 + 16*mt + r + 8*half, col nw + 16*s + l16
#pragma unroll
    for (int mt = 0; mt < 2; ++mt) {
#pragma unroll
        for (int s = 0; s < 4; ++s) {
            int col = nw + s * 16 + l16;
            float bi = bias ? bias[col] : 0.0f;
#pragma unroll
            for (int r = 0; r < 8; ++r) {
                int row = m0 + mt * 16 + r + 8 * half;
                float v = acc[mt][s][r] + bi;
                if (act == 1) v = tanhf(v);
                if (Cf) Cf[(size_t)row * ldcf + col] = v;
                if (Cb) Cb[(size_t)row * ldcb + col] = (bf16_t)v;
            }
        }
    }
}

// ---------------------------------------------------------------------------
// Host orchestration
// ---------------------------------------------------------------------------
static inline size_t alignup(size_t x) { return (x + 255) & ~(size_t)255; }

extern "C" void kernel_launch(void* const* d_in, const int* in_sizes, int n_in,
                              void* d_out, int out_size, void* d_ws, size_t ws_size,
                              hipStream_t stream) {
    // setup_inputs() order
    const int*   target = (const int*)  d_in[0];   // (B, T)
    const float* src_enc= (const float*)d_in[1];   // (B, S, 2H)
    const float* h0     = (const float*)d_in[2];   // (B, H)
    const float* c0     = (const float*)d_in[3];   // (B, H)
    const float* emb    = (const float*)d_in[4];   // (V, E)
    const float* W_ih   = (const float*)d_in[5];   // (E, 4H)
    const float* W_hh   = (const float*)d_in[6];   // (H, 4H)
    const float* b_rnn  = (const float*)d_in[7];   // (4H,)
    const float* W_attn = (const float*)d_in[8];   // (2H, H)
    const float* b_attn = (const float*)d_in[9];   // (H,)
    const float* W_comb = (const float*)d_in[10];  // (2H, H)
    const float* b_comb = (const float*)d_in[11];  // (H,)
    const float* W_out  = (const float*)d_in[12];  // (H, V)
    const float* b_out  = (const float*)d_in[13];  // (V,)
    float* logits = (float*)d_out;                 // (B, T-1, V)

    // ---- workspace carve (transposed bf16 weights) ----
    char* ws = (char*)d_ws;
    size_t off = 0;
    auto carve = [&](size_t bytes) { char* p = ws + off; off += alignup(bytes); return p; };
    bf16_t* W_catT    = (bf16_t*)carve((size_t)G4H * KCAT * 2);   // (4096 x 1536)
    bf16_t* W_combT   = (bf16_t*)carve((size_t)H_ * H2 * 2);      // (1024 x 2048)
    bf16_t* W_outT    = (bf16_t*)carve((size_t)V_ * H_ * 2);      // (32000 x 1024)
    bf16_t* W_attnT   = (bf16_t*)carve((size_t)H_ * H2 * 2);      // (1024 x 2048)
    bf16_t* src_enc_b = (bf16_t*)carve((size_t)B_ * S_ * H2 * 2); // row-major A
    float*  enc_proj  = (float*) carve((size_t)B_ * S_ * H_ * 4);
    bf16_t* xs_b      = (bf16_t*)carve((size_t)B_ * TM1 * E_ * 2);
    float*  h_f       = (float*) carve((size_t)B_ * H_ * 4);
    float*  c_f       = (float*) carve((size_t)B_ * H_ * 4);
    bf16_t* A_cat     = (bf16_t*)carve((size_t)B_ * KCAT * 2);
    bf16_t* hc_b      = (bf16_t*)carve((size_t)B_ * H2 * 2);
    float*  gates_f   = (float*) carve((size_t)B_ * G4H * 4);
    bf16_t* outs_b    = (bf16_t*)carve((size_t)B_ * TM1 * H_ * 2);
    (void)ws_size; (void)in_sizes; (void)n_in; (void)out_size;

    const int TB = 256;
    auto blocks = [](size_t n, int tb) { return (unsigned)((n + tb - 1) / tb); };
    dim3 tpose_blk(32, 8);

    // ---- weight conversions + transposes (f32 -> bf16, B^T layout) ----
    // W_catT = [W_ih; W_hh]^T : rows n in [0,4096), cols k in [0,1536)
    k_f2bT<<<dim3(G4H / 32, E_ / 32), tpose_blk, 0, stream>>>(W_ih, W_catT, E_, G4H, KCAT, 0);
    k_f2bT<<<dim3(G4H / 32, H_ / 32), tpose_blk, 0, stream>>>(W_hh, W_catT, H_, G4H, KCAT, E_);
    k_f2bT<<<dim3(H_ / 32, H2 / 32), tpose_blk, 0, stream>>>(W_comb, W_combT, H2, H_, H2, 0);
    k_f2bT<<<dim3(V_ / 32, H_ / 32), tpose_blk, 0, stream>>>(W_out, W_outT, H_, V_, H_, 0);
    k_f2bT<<<dim3(H_ / 32, H2 / 32), tpose_blk, 0, stream>>>(W_attn, W_attnT, H2, H_, H2, 0);
    k_f2b<<<blocks((size_t)B_ * S_ * H2, TB), TB, 0, stream>>>(src_enc, src_enc_b, B_ * S_ * H2);
    k_gather_emb<<<blocks((size_t)B_ * TM1 * E_, TB), TB, 0, stream>>>(target, emb, xs_b, B_ * TM1 * E_);

    // ---- init recurrent state ----
    hipMemcpyAsync(h_f, h0, (size_t)B_ * H_ * 4, hipMemcpyDeviceToDevice, stream);
    hipMemcpyAsync(c_f, c0, (size_t)B_ * H_ * 4, hipMemcpyDeviceToDevice, stream);

    // ---- enc_proj = src_enc @ W_attn + b_attn : M=2048, N=1024, K=2048 ----
    k_gemm_wmma<<<dim3(H_ / 256, (B_ * S_) / 32), 128, 0, stream>>>(
        src_enc_b, H2, W_attnT, b_attn, enc_proj, H_, nullptr, 0, H2, 0);

    // ---- decode loop ----
    for (int t = 0; t < TM1; ++t) {
        k_assemble<<<blocks((size_t)B_ * KCAT, TB), TB, 0, stream>>>(xs_b, t, h_f, A_cat);
        // gates = [x_t|h] @ W_cat + b_rnn : M=32, N=4096, K=1536
        k_gemm_wmma<<<dim3(G4H / 256, 1), 128, 0, stream>>>(
            A_cat, KCAT, W_catT, b_rnn, gates_f, G4H, nullptr, 0, KCAT, 0);
        k_lstm<<<blocks((size_t)B_ * H_, TB), TB, 0, stream>>>(gates_f, c_f, h_f, hc_b);
        k_attn<<<B_, TB, 0, stream>>>(enc_proj, h_f, hc_b);
        // out_t = tanh([h|ctx] @ W_comb + b_comb) -> outs[:, t, :] (bf16) : M=32, N=1024, K=2048
        k_gemm_wmma<<<dim3(H_ / 256, 1), 128, 0, stream>>>(
            hc_b, H2, W_combT, b_comb, nullptr, 0, outs_b + (size_t)t * H_, TM1 * H_, H2, 1);
    }

    // ---- logits = outs @ W_out + b_out : M=2016, N=32000, K=1024 ----
    k_gemm_wmma<<<dim3(V_ / 256, (B_ * TM1) / 32), 128, 0, stream>>>(
        outs_b, H_, W_outT, b_out, logits, V_, nullptr, 0, H_, 0);
}
